// Mamba2_45913200394711
// MI455X (gfx1250) — matrix operand order repaired and checked by reading, not compiled, not run on hardware
//
#include <hip/hip_runtime.h>
#include <hip/hip_bf16.h>
#include <math.h>

// ---------------- problem constants (from reference) ----------------
#define B_      2
#define L_      512
#define D_MODEL 512
#define D_INNER 1024
#define NHEADS  16
#define HEAD_P  64
#define NSTATE  64
#define CONV_DIM 1152           // D_INNER + 2*64
#define NPROJ   2192            // 2*D_INNER + 2*64 + NHEADS
#define BL      (B_ * L_)       // 1024 token rows
#define DT_OFF  (D_INNER + CONV_DIM)   // 2176: dt columns in zxbcdt
#define RMS_EPS 1e-5f

// ---------------- WMMA types ----------------
typedef __attribute__((ext_vector_type(16))) __bf16 v16bf;
typedef __attribute__((ext_vector_type(8)))  __bf16 bf16x8;
typedef __attribute__((ext_vector_type(8)))  float  v8f;

// Pack 2 f32 -> 2 bf16 in one dword: bit-round (nearest, ties-away) with a
// +0x8000 add, then one v_perm_b32 grabs both high halves.
// __builtin_amdgcn_perm(s0, s1, sel): sel byte 0-3 -> byte of s1, 4-7 -> byte of s0.
__device__ inline unsigned pack_bf16_pair(float f0, float f1) {
    unsigned u0 = __float_as_uint(f0) + 0x8000u;
    unsigned u1 = __float_as_uint(f1) + 0x8000u;
    return __builtin_amdgcn_perm(u1, u0, 0x07060302u);  // {u1[31:16], u0[31:16]}
}
// 8 floats -> 16B of bf16 (one ds_store_b128)
__device__ inline uint4 pack8(float f0, float f1, float f2, float f3,
                              float f4, float f5, float f6, float f7) {
    uint4 r;
    r.x = pack_bf16_pair(f0, f1);
    r.y = pack_bf16_pair(f2, f3);
    r.z = pack_bf16_pair(f4, f5);
    r.w = pack_bf16_pair(f6, f7);
    return r;
}

__device__ inline float siluf(float x)     { return x / (1.0f + __expf(-x)); }
__device__ inline float softplusf(float x) { return (x > 20.0f) ? x : log1pf(__expf(x)); }

// =====================================================================
// f32-in / f32-out GEMM via bf16 WMMA, f32 accumulate.
// Block = 256 thr (8 waves, 2x4 wave grid), tile 64(M) x 128(N), K-step 32.
// Requirements (met by both call sites): M % 64 == 0, K % 32 == 0.
// N arbitrary: staging clamps addresses (garbage columns feed only dead
// output columns, which guarded stores skip).
// Double-buffered LDS software pipeline: global prefetch overlaps WMMA.
// =====================================================================
#define BM 64
#define BN 128
#define BK 32
#define LDSR (BK + 16)   // 48 bf16 = 96 B row stride (rows stay 16B aligned)

// A-fragment (16x32 bf16): lane<16 -> M=lane, K=0..15;
// lane>=16 -> same M, K=16..31
__device__ inline v16bf load_a_frag(const __bf16* row, int half) {
    int kb = half * 16;
    bf16x8 lo = *(const bf16x8*)(row + kb);
    bf16x8 hi = *(const bf16x8*)(row + kb + 8);
    v16bf r;
#pragma unroll
    for (int i = 0; i < 8; ++i) { r[i] = lo[i]; r[8 + i] = hi[i]; }
    return r;
}
// B-fragment (32x16 bf16): lane&15 = N; lanes<16 K=0..15, lanes>=16 K=16..31.
// LDS holds B transposed (N-major) -> 2 contiguous b128 loads.
__device__ inline v16bf load_b_frag(const __bf16* row, int half) {
    int kb = half * 16;
    bf16x8 lo = *(const bf16x8*)(row + kb);
    bf16x8 hi = *(const bf16x8*)(row + kb + 8);
    v16bf r;
#pragma unroll
    for (int i = 0; i < 8; ++i) { r[i] = lo[i]; r[8 + i] = hi[i]; }
    return r;
}
// C/D layout: VGPR r <-> M = r + 8*(lane>=16); N = lane&15
__device__ inline void store_frag(float* __restrict__ C, v8f c,
                                  int m0, int n0, int lane, int N) {
    int n  = n0 + (lane & 15);
    int mb = m0 + ((lane >> 4) * 8);
    if (n >= N) return;
#pragma unroll
    for (int r = 0; r < 8; ++r)
        C[(size_t)(mb + r) * N + n] = c[r];
}

__global__ __launch_bounds__(256)
void gemm_bf16_wmma(const float* __restrict__ A, const float* __restrict__ Bg,
                    float* __restrict__ C, int M, int N, int K) {
    __shared__ __bf16 As[2][BM][LDSR];
    __shared__ __bf16 Bs[2][BN][LDSR];   // transposed: [n][k]

    const int tid  = threadIdx.x;
    const int lane = tid & 31;
    const int wave = tid >> 5;
    const int wrow = wave >> 2;       // 0..1
    const int wcol = wave & 3;        // 0..3
    const int blockM = blockIdx.y * BM;
    const int blockN = blockIdx.x * BN;
    const int half = lane >> 4;
    const int mr   = lane & 15;

    // ---- staging thread mappings (branch-free) ----
    // A: r = tid>>2 (0..63 rows), cb = (tid&3)*8 -> two float4 global loads,
    //    one b128 LDS store.
    const int ar = tid >> 2;
    const int ac = (tid & 3) * 8;
    // B: n = tid&127 (lane-coalesced global b32 loads), kg = tid>>7 (0..1),
    //    each thread owns 16 consecutive k -> two b128 LDS stores.
    const int bn = tid & 127;
    const int bkg = (tid >> 7) * 16;
    const int nclamp = (blockN + bn < N) ? (blockN + bn) : (N - 1);

    const float* Abase = A + (size_t)(blockM + ar) * K + ac;

    v8f c00 = {}; v8f c01 = {}; v8f c10 = {}; v8f c11 = {};

    // ---- prologue: stage tile 0 into buffer 0 ----
    {
        float4 a0 = *(const float4*)(Abase);
        float4 a1 = *(const float4*)(Abase + 4);
        *(uint4*)&As[0][ar][ac] =
            pack8(a0.x, a0.y, a0.z, a0.w, a1.x, a1.y, a1.z, a1.w);

        float bv[16];
#pragma unroll
        for (int j = 0; j < 16; ++j)
            bv[j] = Bg[(size_t)(bkg + j) * N + nclamp];
        *(uint4*)&Bs[0][bn][bkg] =
            pack8(bv[0], bv[1], bv[2], bv[3], bv[4], bv[5], bv[6], bv[7]);
        *(uint4*)&Bs[0][bn][bkg + 8] =
            pack8(bv[8], bv[9], bv[10], bv[11], bv[12], bv[13], bv[14], bv[15]);
    }

    int buf = 0;
    for (int k0 = 0; k0 < K; k0 += BK) {
        __syncthreads();   // current buffer visible; other buffer free to overwrite

        const bool has_next = (k0 + BK) < K;
        float4 a0n, a1n;
        float  bvn[16];
        if (has_next) {                              // uniform branch
            const float* Ap = Abase + (k0 + BK);
            a0n = *(const float4*)(Ap);
            a1n = *(const float4*)(Ap + 4);
            const float* Bp = Bg + (size_t)(k0 + BK + bkg) * N + nclamp;
#pragma unroll
            for (int j = 0; j < 16; ++j)
                bvn[j] = Bp[(size_t)j * N];
        }

        // ---- compute on current buffer ----
        const int am0 = wrow * 32;
        const int bn0 = wcol * 32;
        v16bf a0 = load_a_frag(&As[buf][am0 + mr][0],      half);
        v16bf a1 = load_a_frag(&As[buf][am0 + 16 + mr][0], half);
        v16bf b0 = load_b_frag(&Bs[buf][bn0 + mr][0],      half);
        v16bf b1 = load_b_frag(&Bs[buf][bn0 + 16 + mr][0], half);

        c00 = __builtin_amdgcn_wmma_f32_16x16x32_bf16(false, a0, false, b0, (short)0, c00, false, false);
        c01 = __builtin_amdgcn_wmma_f32_16x16x32_bf16(false, a0, false, b1, (short)0, c01, false, false);
        c10 = __builtin_amdgcn_wmma_f32_16x16x32_bf16(false, a1, false, b0, (short)0, c10, false, false);
        c11 = __builtin_amdgcn_wmma_f32_16x16x32_bf16(false, a1, false, b1, (short)0, c11, false, false);

        // ---- write prefetched tile into the other buffer ----
        if (has_next) {
            int nb = buf ^ 1;
            *(uint4*)&As[nb][ar][ac] =
                pack8(a0n.x, a0n.y, a0n.z, a0n.w, a1n.x, a1n.y, a1n.z, a1n.w);
            *(uint4*)&Bs[nb][bn][bkg] =
                pack8(bvn[0], bvn[1], bvn[2], bvn[3], bvn[4], bvn[5], bvn[6], bvn[7]);
            *(uint4*)&Bs[nb][bn][bkg + 8] =
                pack8(bvn[8], bvn[9], bvn[10], bvn[11], bvn[12], bvn[13], bvn[14], bvn[15]);
        }
        buf ^= 1;
    }

    const int m0 = blockM + wrow * 32;
    const int n0 = blockN + wcol * 32;
    store_frag(C, c00, m0,      n0,      lane, N);
    store_frag(C, c01, m0,      n0 + 16, lane, N);
    store_frag(C, c10, m0 + 16, n0,      lane, N);
    store_frag(C, c11, m0 + 16, n0 + 16, lane, N);
}

// =====================================================================
// Prep: causal depthwise conv(w=3) + bias + SiLU over xBC channels of
// zxbcdt; split into x_raw, x*dt (dt = softplus(dt_raw+bias)), B, C.
// =====================================================================
__global__ __launch_bounds__(256)
void prep_kernel(const float* __restrict__ zx,       // [BL, NPROJ]
                 const float* __restrict__ conv_w,   // [CONV_DIM, 3]
                 const float* __restrict__ conv_b,   // [CONV_DIM]
                 const float* __restrict__ dt_bias,  // [NHEADS]
                 float* __restrict__ xraw,           // [BL, D_INNER]
                 float* __restrict__ xs,             // [BL, D_INNER]
                 float* __restrict__ Bv,             // [BL, NSTATE]
                 float* __restrict__ Cv) {           // [BL, NSTATE]
    int gid = blockIdx.x * blockDim.x + threadIdx.x;
    if (gid >= BL * CONV_DIM) return;
    int pos = gid / CONV_DIM;     // b*L + l
    int c   = gid - pos * CONV_DIM;
    int l   = pos & (L_ - 1);

    const float w0 = conv_w[c * 3 + 0];
    const float w1 = conv_w[c * 3 + 1];
    const float w2 = conv_w[c * 3 + 2];

    const size_t base = (size_t)pos * NPROJ + D_INNER + c;   // xBC(b,l,c)
    float acc = zx[base] * w2;
    if (l >= 1) acc += zx[base - NPROJ]     * w1;
    if (l >= 2) acc += zx[base - 2 * NPROJ] * w0;
    float v = siluf(acc + conv_b[c]);

    if (c < D_INNER) {
        int h = c >> 6;
        float dtv = softplusf(zx[(size_t)pos * NPROJ + DT_OFF + h] + dt_bias[h]);
        xraw[(size_t)pos * D_INNER + c] = v;
        xs  [(size_t)pos * D_INNER + c] = v * dtv;
    } else if (c < D_INNER + NSTATE) {
        Bv[(size_t)pos * NSTATE + (c - D_INNER)] = v;
    } else {
        Cv[(size_t)pos * NSTATE + (c - D_INNER - NSTATE)] = v;
    }
}

// =====================================================================
// Selective scan (SSD with chunk==1 collapses to classic recurrence):
//   h_t = exp(A*dt_t) * h_{t-1} + B_t (x_t*dt_t)^T ;  y_t = C_t . h_t
// One block per (batch, head); 64x64 state in registers (256 thr x 16).
// =====================================================================
#define TC 16
__global__ __launch_bounds__(256)
void scan_kernel(const float* __restrict__ zx,      // [BL, NPROJ] (dt cols)
                 const float* __restrict__ xs,      // [BL, D_INNER]
                 const float* __restrict__ xraw,    // [BL, D_INNER]
                 const float* __restrict__ Bv,      // [BL, NSTATE]
                 const float* __restrict__ Cv,      // [BL, NSTATE]
                 const float* __restrict__ dt_bias, // [NHEADS]
                 const float* __restrict__ A_log,   // [NHEADS]
                 const float* __restrict__ D_param, // [NHEADS]
                 float* __restrict__ y) {           // [BL, D_INNER]
    __shared__ float sB[TC][NSTATE];
    __shared__ float sC[TC][NSTATE];
    __shared__ float sX[TC][HEAD_P];
    __shared__ float sA[TC];

    const int b   = blockIdx.x >> 4;
    const int h   = blockIdx.x & 15;
    const int tid = threadIdx.x;
    const int p   = tid >> 2;            // 0..63
    const int nb  = (tid & 3) * 16;      // n slice base
    const float negA = -__expf(A_log[h]);
    const float Dp   = D_param[h];

    float hreg[16];
#pragma unroll
    for (int i = 0; i < 16; ++i) hreg[i] = 0.0f;

    for (int tbase = 0; tbase < L_; tbase += TC) {
        __syncthreads();   // previous chunk compute done
        for (int idx = tid; idx < TC * NSTATE; idx += 256) {
            int t = idx >> 6, n = idx & 63;
            size_t pos = (size_t)(b * L_ + tbase + t);
            sB[t][n] = Bv[pos * NSTATE + n];
            sC[t][n] = Cv[pos * NSTATE + n];
            sX[t][n] = xs[pos * D_INNER + h * HEAD_P + n];
        }
        if (tid < TC) {
            size_t pos = (size_t)(b * L_ + tbase + tid);
            float dtv = softplusf(zx[pos * NPROJ + DT_OFF + h] + dt_bias[h]);
            sA[tid] = __expf(negA * dtv);
        }
        __syncthreads();

#pragma unroll
        for (int t = 0; t < TC; ++t) {
            const float a    = sA[t];
            const float xval = sX[t][p];
            float acc = 0.0f;
#pragma unroll
            for (int i = 0; i < 16; ++i) {
                float hn = a * hreg[i] + sB[t][nb + i] * xval;
                hreg[i] = hn;
                acc += sC[t][nb + i] * hn;
            }
            acc += __shfl_xor(acc, 1, 32);
            acc += __shfl_xor(acc, 2, 32);
            if ((tid & 3) == 0) {
                size_t pos = (size_t)(b * L_ + tbase + t);
                y[pos * D_INNER + h * HEAD_P + p] =
                    acc + xraw[pos * D_INNER + h * HEAD_P + p] * Dp;
            }
        }
    }
}

// =====================================================================
// Gating + RMSNorm:  yz = y * silu(z);  out = yz * rsqrt(mean(yz^2)+eps) * w
// =====================================================================
__global__ __launch_bounds__(256)
void rmsnorm_kernel(const float* __restrict__ y,     // [BL, D_INNER]
                    const float* __restrict__ zx,    // [BL, NPROJ] (z = first 1024)
                    const float* __restrict__ rms_w, // [D_INNER]
                    float* __restrict__ out) {       // [BL, D_INNER]
    __shared__ float red[256];
    const int row = blockIdx.x;
    const int tid = threadIdx.x;
    float yzv[4];
    float ss = 0.0f;
#pragma unroll
    for (int i = 0; i < 4; ++i) {
        int c = tid + i * 256;
        float yv = y[(size_t)row * D_INNER + c];
        float zv = zx[(size_t)row * NPROJ + c];
        float yz = yv * siluf(zv);
        yzv[i] = yz;
        ss += yz * yz;
    }
    red[tid] = ss;
    __syncthreads();
    for (int s = 128; s > 0; s >>= 1) {
        if (tid < s) red[tid] += red[tid + s];
        __syncthreads();
    }
    float scale = rsqrtf(red[0] * (1.0f / D_INNER) + RMS_EPS);
#pragma unroll
    for (int i = 0; i < 4; ++i) {
        int c = tid + i * 256;
        out[(size_t)row * D_INNER + c] = yzv[i] * scale * rms_w[c];
    }
}

// =====================================================================
// Host-side orchestration
// =====================================================================
extern "C" void kernel_launch(void* const* d_in, const int* in_sizes, int n_in,
                              void* d_out, int out_size, void* d_ws, size_t ws_size,
                              hipStream_t stream) {
    (void)in_sizes; (void)n_in; (void)out_size; (void)ws_size;
    const float* u       = (const float*)d_in[0];   // [B,L,512]
    const float* W_in    = (const float*)d_in[1];   // [512, 2192]
    const float* conv_w  = (const float*)d_in[2];   // [1152, 3]
    const float* conv_b  = (const float*)d_in[3];   // [1152]
    const float* dt_bias = (const float*)d_in[4];   // [16]
    const float* A_log   = (const float*)d_in[5];   // [16]
    const float* D_param = (const float*)d_in[6];   // [16]
    const float* rms_w   = (const float*)d_in[7];   // [1024]
    const float* W_out   = (const float*)d_in[8];   // [1024, 512]
    float* out = (float*)d_out;                     // [B,L,512] f32

    // workspace layout (floats)
    float* ws = (float*)d_ws;
    size_t off = 0;
    float* zx    = ws + off; off += (size_t)BL * NPROJ;    // 2,244,608
    float* xraw  = ws + off; off += (size_t)BL * D_INNER;  // 1,048,576
    float* xs    = ws + off; off += (size_t)BL * D_INNER;  // 1,048,576
    float* Bv    = ws + off; off += (size_t)BL * NSTATE;   //    65,536
    float* Cv    = ws + off; off += (size_t)BL * NSTATE;   //    65,536
    float* yssd  = ws + off; off += (size_t)BL * D_INNER;  // 1,048,576
    float* ynorm = xs;  // xs dead after scan; reuse for normed activations

    // 1) in_proj: zx[1024,2192] = u[1024,512] @ W_in[512,2192]
    {
        dim3 grid((NPROJ + BN - 1) / BN, BL / BM);
        gemm_bf16_wmma<<<grid, 256, 0, stream>>>(u, W_in, zx, BL, NPROJ, D_MODEL);
    }
    // 2) conv + silu + split + dt scaling
    {
        int total = BL * CONV_DIM;
        prep_kernel<<<(total + 255) / 256, 256, 0, stream>>>(
            zx, conv_w, conv_b, dt_bias, xraw, xs, Bv, Cv);
    }
    // 3) selective scan (one block per batch*head)
    scan_kernel<<<B_ * NHEADS, 256, 0, stream>>>(
        zx, xs, xraw, Bv, Cv, dt_bias, A_log, D_param, yssd);
    // 4) gating + RMSNorm
    rmsnorm_kernel<<<BL, 256, 0, stream>>>(yssd, zx, rms_w, ynorm);
    // 5) out_proj: out[1024,512] = ynorm[1024,1024] @ W_out[1024,512]
    {
        dim3 grid((D_MODEL + BN - 1) / BN, BL / BM);
        gemm_bf16_wmma<<<grid, 256, 0, stream>>>(ynorm, W_out, out, BL, D_MODEL, D_INNER);
    }
}
